// ContractExpand_76794015252629
// MI455X (gfx1250) — matrix operand order repaired and checked by reading, not compiled, not run on hardware
//
#include <hip/hip_runtime.h>
#include <hip/hip_bf16.h>

// ---------------- problem constants (from reference) ----------------
#define DIMC   300          // feature dim D
#define KPAD   320          // D padded up to a multiple of 32 (K of bf16 WMMA)
#define NPADR  320          // W rows padded to 20 full 16-wide N tiles (10 pairs)
#define BATCH  64
#define LSEQ   800

typedef __attribute__((ext_vector_type(16))) __bf16 v16bf;
typedef __attribute__((ext_vector_type(8)))  __bf16 v8bf;
typedef __attribute__((ext_vector_type(8)))  float  v8f;

__device__ __forceinline__ __bf16 f2bf(float f) {
    // round-to-nearest-even f32 -> bf16 via bit manipulation
    union { float f; unsigned u; } in; in.f = f;
    unsigned r = in.u + 0x7FFFu + ((in.u >> 16) & 1u);
    union { unsigned short s; __bf16 b; } out; out.s = (unsigned short)(r >> 16);
    return out.b;
}

// ---------------- kernel 1: per-scale segment sum -> bf16 (K-padded) ----------------
// one block (320 threads) per seg row; seg row = sum of R consecutive x rows
template<int R, int G>
__global__ void segsum_kernel(const float* __restrict__ x, __bf16* __restrict__ seg) {
    const int row = blockIdx.x;            // 0 .. B*G-1  (= b*G + g)
    const int b   = row / G;
    const int g   = row % G;
    const int k   = threadIdx.x;           // 0 .. 319
    float s = 0.f;
    if (k < DIMC) {
        const float* p = x + ((size_t)b * LSEQ + (size_t)g * R) * DIMC + k;
        #pragma unroll
        for (int j = 0; j < R; ++j) s += p[(size_t)j * DIMC];
    }
    seg[(size_t)row * KPAD + k] = f2bf(s);  // k in [300,320) stores 0
}

// ---------------- kernel 2: W (5 x 300 x 300, y = seg @ W^T) -> bf16, padded ----------------
__global__ void wconv_kernel(const float* __restrict__ W, __bf16* __restrict__ Wb) {
    const int row = blockIdx.x;            // 0 .. 5*NPADR-1
    const int s   = row / NPADR;
    const int e   = row % NPADR;           // output-feature index (W row)
    const int k   = threadIdx.x;           // 0 .. 319
    float v = 0.f;
    if (e < DIMC && k < DIMC) v = W[((size_t)s * DIMC + e) * DIMC + k];
    Wb[(size_t)row * KPAD + k] = f2bf(v);
}

// ---------------- kernel 3: WMMA GEMM (32x32 per wave) + bias + relu + 1/r + expand ----------
// blockDim = 64 (2 waves); each wave owns a 32(M)x32(N) register tile = 2x2 WMMA tiles:
//   M rows : blockIdx.y*32 .. +31   (row = b*G + g within this scale)
//   N cols : (blockIdx.x*2 + waveId)*32 .. +31  over the padded 320 output features
// A fragments are shared across the two N tiles, B fragments across the two M tiles:
// 8 b128 loads feed 4 WMMAs (2 loads/WMMA instead of 4).
template<int R, int G>
__global__ void gemm_expand_kernel(const __bf16* __restrict__ seg,  // [B*G][KPAD]
                                   const __bf16* __restrict__ Wb,   // [NPADR][KPAD]
                                   const float*  __restrict__ bias, // [DIMC]
                                   float*        __restrict__ out)  // [B][LSEQ][DIMC]
{
    const int wave    = threadIdx.x >> 5;
    const int ln      = threadIdx.x & 31;
    const int npair   = blockIdx.x * 2 + wave;   // 0..9 exactly (grid.x=5, 2 waves)
    const int mpair   = blockIdx.y;
    const int lane16  = ln & 15;
    const int half    = ln >> 4;                 // K-half selector (contiguous K order used here)

    // A: lane holds row; per 32-K step it owns a contiguous K in [half*16,+16)
    const __bf16* aBase0 = seg + (size_t)(mpair * 32 + lane16)      * KPAD + half * 16;
    const __bf16* aBase1 = seg + (size_t)(mpair * 32 + 16 + lane16) * KPAD + half * 16;
    // B: lane holds column of W^T == W row; per 32-K step it owns a contiguous
    //    32B chunk K in [half*16, +16)
    const __bf16* bBase0 = Wb + (size_t)(npair * 32 + lane16)      * KPAD + half * 16;
    const __bf16* bBase1 = Wb + (size_t)(npair * 32 + 16 + lane16) * KPAD + half * 16;

    v8f acc00 = {}, acc01 = {}, acc10 = {}, acc11 = {};
    #pragma unroll
    for (int it = 0; it < KPAD / 32; ++it) {
        union { v16bf v; v8bf h[2]; } A0, A1;
        A0.h[0] = *(const v8bf*)(aBase0 + it * 32);
        A0.h[1] = *(const v8bf*)(aBase0 + it * 32 + 8);
        A1.h[0] = *(const v8bf*)(aBase1 + it * 32);
        A1.h[1] = *(const v8bf*)(aBase1 + it * 32 + 8);
        v16bf B0 = *(const v16bf*)(bBase0 + it * 32);
        v16bf B1 = *(const v16bf*)(bBase1 + it * 32);
        acc00 = __builtin_amdgcn_wmma_f32_16x16x32_bf16(false, A0.v, false, B0, (short)0, acc00, false, false);
        acc01 = __builtin_amdgcn_wmma_f32_16x16x32_bf16(false, A0.v, false, B1, (short)0, acc01, false, false);
        acc10 = __builtin_amdgcn_wmma_f32_16x16x32_bf16(false, A1.v, false, B0, (short)0, acc10, false, false);
        acc11 = __builtin_amdgcn_wmma_f32_16x16x32_bf16(false, A1.v, false, B1, (short)0, acc11, false, false);
    }

    // C/D layout: lane -> column n = lane16, VGPR v -> row m = v + 8*half
    const float inv_r = 1.0f / (float)R;
    #pragma unroll
    for (int tj = 0; tj < 2; ++tj) {
        const int  d      = (npair * 2 + tj) * 16 + lane16;
        const bool dvalid = (d < DIMC);
        const float bv    = dvalid ? bias[d] : 0.f;
        #pragma unroll
        for (int ti = 0; ti < 2; ++ti) {
            const v8f& acc = (ti == 0) ? (tj == 0 ? acc00 : acc01)
                                       : (tj == 0 ? acc10 : acc11);
            #pragma unroll
            for (int v = 0; v < 8; ++v) {
                const int grow = mpair * 32 + ti * 16 + v + 8 * half;  // = b*G + g
                const int bidx = grow / G;
                const int gidx = grow % G;
                float val = fmaxf(acc[v] + bv, 0.f) * inv_r;
                if (dvalid) {
                    float* o = out + ((size_t)bidx * LSEQ + (size_t)gidx * R) * DIMC + d;
                    #pragma unroll
                    for (int j = 0; j < R; ++j) o[(size_t)j * DIMC] = val;  // repeat-expand
                }
            }
        }
    }
}

// ---------------- host-side launch ----------------
template<int S, int R, int G>
static void launch_scale(const float* x, __bf16* segAll, const __bf16* Wb,
                         const float* bias, float* out, size_t rowBase,
                         hipStream_t stream)
{
    segsum_kernel<R, G><<<dim3(BATCH * G), dim3(KPAD), 0, stream>>>(
        x, segAll + rowBase * KPAD);
    gemm_expand_kernel<R, G><<<dim3(5, (BATCH * G) / 32), dim3(64), 0, stream>>>(
        segAll + rowBase * KPAD,
        Wb + (size_t)S * NPADR * KPAD,
        bias + (size_t)S * DIMC,
        out + (size_t)S * BATCH * LSEQ * DIMC);
}

extern "C" void kernel_launch(void* const* d_in, const int* in_sizes, int n_in,
                              void* d_out, int out_size, void* d_ws, size_t ws_size,
                              hipStream_t stream) {
    (void)in_sizes; (void)n_in; (void)out_size; (void)ws_size;
    const float* x    = (const float*)d_in[0];   // [64][800][300] f32
    const float* W    = (const float*)d_in[1];   // [5][300][300]  f32
    const float* bias = (const float*)d_in[2];   // [5][300]       f32
    float* out        = (float*)d_out;           // [5][64][800][300] f32

    // workspace: seg bf16 [96768][320] (61.9 MB) then W bf16 [5*320][320] (~1 MB)
    __bf16* seg = (__bf16*)d_ws;
    const size_t SEG_ELEMS = 96768ull * KPAD;
    __bf16* Wb  = seg + SEG_ELEMS;

    wconv_kernel<<<dim3(5 * NPADR), dim3(KPAD), 0, stream>>>(W, Wb);

    // per-scale row bases in the concatenated seg buffer (all multiples of 32)
    launch_scale<0,  1, 800>(x, seg, Wb, bias, out,     0, stream);
    launch_scale<1,  2, 400>(x, seg, Wb, bias, out, 51200, stream);
    launch_scale<2,  4, 200>(x, seg, Wb, bias, out, 76800, stream);
    launch_scale<3, 10,  80>(x, seg, Wb, bias, out, 89600, stream);
    launch_scale<4, 25,  32>(x, seg, Wb, bias, out, 94720, stream);
}
